// GraspNet_33397665693877
// MI455X (gfx1250) — matrix operand order repaired and checked by reading, not compiled, not run on hardware
//
#include <hip/hip_runtime.h>

#define NPOINT  1024
#define NSAMPLE 64
#define NPTS    4096
#define NBATCH  4
#define CFEAT   256
#define HMIN_F  (-0.02f)
#define HMAX_F  (0.04f)

typedef __attribute__((ext_vector_type(16))) __bf16 v16bf;
typedef __attribute__((ext_vector_type(8)))  float  v8f;
typedef unsigned int u32x4 __attribute__((ext_vector_type(4)));

union FragU { u32x4 q[2]; v16bf v; };

__device__ __forceinline__ unsigned short f2bf(float f) {
  unsigned u = __builtin_bit_cast(unsigned, f);
  u += 0x7FFFu + ((u >> 16) & 1u);          // round-to-nearest-even
  return (unsigned short)(u >> 16);
}
__device__ __forceinline__ unsigned pack2(float a, float b) {
  return (unsigned)f2bf(a) | ((unsigned)f2bf(b) << 16);
}

// A fragment: 16x32 bf16 from row-major [lda] storage.
// Lane L (0-15): M=L, K pairs {0,1..6,7 | 16..23}; lane L+16: K pairs {8..15 | 24..31}.
__device__ __forceinline__ v16bf load_a_frag(const unsigned short* __restrict__ A,
                                             int lda, int rowBase, int kBase, int lane) {
  const int r  = lane & 15;
  const int hi = lane >> 4;                  // 0 or 1 -> k-base 0 or 8 (=4 dwords)
  const u32x4* p = (const u32x4*)(A + (size_t)(rowBase + r) * lda + kBase);
  FragU f;
  f.q[0] = p[hi];                            // dwords kb/2 .. kb/2+3
  f.q[1] = p[hi + 2];                        // dwords kb/2+8 .. kb/2+11
  return f.v;
}

// B fragment: 32x16 bf16; source stored sample-major Bt[n][k] (ld = ldk) so the
// (K=2i,2i+1) pair per VGPR is one contiguous dword. Lanes 0-15: K 0-7,16-23 (N=lane);
// lanes 16-31: K 8-15,24-31 (N=lane-16).
__device__ __forceinline__ v16bf load_b_frag(const unsigned short* Bt,
                                             int ldk, int nBase, int kBase, int lane) {
  const int n  = lane & 15;
  const int hi = lane >> 4;
  const u32x4* p = (const u32x4*)(Bt + (size_t)(nBase + n) * ldk + kBase);
  FragU f;
  f.q[0] = p[hi];
  f.q[1] = p[hi + 2];
  return f.v;
}

// ---------------------------------------------------------------------------
// Kernel 0: convert W1 (256x259 f32, zero-pad K to 288) and W2 (256x256) to bf16
// ---------------------------------------------------------------------------
__global__ void convert_weights_kernel(const float* __restrict__ W1,
                                       const float* __restrict__ W2,
                                       unsigned short* __restrict__ W1b,
                                       unsigned short* __restrict__ W2b) {
  int i = blockIdx.x * blockDim.x + threadIdx.x;
  if (i < 256 * 288) {
    int r = i / 288, k = i - r * 288;
    float v = (k < 259) ? W1[r * 259 + k] : 0.0f;
    W1b[i] = f2bf(v);
  }
  if (i < 256 * 256) {
    W2b[i] = f2bf(W2[i]);
  }
}

// ---------------------------------------------------------------------------
// Kernel 1: farthest point sampling, one block per batch
// ---------------------------------------------------------------------------
__global__ void __launch_bounds__(256) fps_kernel(const float* __restrict__ xyz,
                                                  int* __restrict__ fps_idx,
                                                  float* __restrict__ new_xyz) {
  __shared__ float sx[NPTS], sy[NPTS], sz[NPTS];
  __shared__ float rv[256];
  __shared__ int   ri[256];
  const int b   = blockIdx.x;
  const int tid = threadIdx.x;
  const float* base = xyz + (size_t)b * NPTS * 3;
  for (int i = tid; i < NPTS; i += 256) {
    sx[i] = base[3 * i + 0];
    sy[i] = base[3 * i + 1];
    sz[i] = base[3 * i + 2];
  }
  float dist[16];
#pragma unroll
  for (int j = 0; j < 16; ++j) dist[j] = 1e10f;
  __syncthreads();

  int cur = 0;
  for (int it = 0; it < NPOINT; ++it) {
    if (tid == 0) {
      fps_idx[b * NPOINT + it] = cur;
      float* nx = new_xyz + (size_t)(b * NPOINT + it) * 3;
      nx[0] = sx[cur]; nx[1] = sy[cur]; nx[2] = sz[cur];
    }
    const float cx = sx[cur], cy = sy[cur], cz = sz[cur];
    float bv = -1.0f; int bi = 0;
#pragma unroll
    for (int j = 0; j < 16; ++j) {
      const int i = tid + j * 256;
      const float dx = sx[i] - cx, dy = sy[i] - cy, dz = sz[i] - cz;
      const float d  = dx * dx + dy * dy + dz * dz;
      const float nd = fminf(dist[j], d);
      dist[j] = nd;
      if (nd > bv) { bv = nd; bi = i; }   // strict > keeps smallest index per thread
    }
    rv[tid] = bv; ri[tid] = bi;
    __syncthreads();
    for (int s = 128; s > 0; s >>= 1) {
      if (tid < s) {
        const float v2 = rv[tid + s]; const int i2 = ri[tid + s];
        if (v2 > rv[tid] || (v2 == rv[tid] && i2 < ri[tid])) { rv[tid] = v2; ri[tid] = i2; }
      }
      __syncthreads();
    }
    cur = ri[0];
    __syncthreads();
  }
}

// ---------------------------------------------------------------------------
// Kernel 2: fused cylinder query + grouping + 2-layer pointwise MLP + max-pool
// One block (256 threads = 8 wave32) per (b, p).
// ---------------------------------------------------------------------------
__global__ void __launch_bounds__(256) grasp_mlp_kernel(
    const float* __restrict__ xyz, const float* __restrict__ features,
    const float* __restrict__ rot, const float* __restrict__ b1,
    const float* __restrict__ b2, const float* __restrict__ new_xyz,
    const unsigned short* __restrict__ W1b, const unsigned short* __restrict__ W2b,
    float* __restrict__ out) {
  __shared__ __align__(16) unsigned short Xt[NSAMPLE][288];   // X^T: [sample][channel], bf16
  __shared__ __align__(16) unsigned short H1t[NSAMPLE][256];  // H1^T, bf16
  __shared__ int   s_idx[NSAMPLE];
  __shared__ float sR[9];
  __shared__ float sC[3];

  const int tid = threadIdx.x;
  const int b   = blockIdx.x >> 10;
  const int p   = blockIdx.x & 1023;

  if (tid < 9) sR[tid] = rot[(size_t)(b * NPOINT + p) * 9 + tid];
  if (tid >= 16 && tid < 19) sC[tid - 16] = new_xyz[(size_t)(b * NPOINT + p) * 3 + (tid - 16)];
  if (tid < NSAMPLE) s_idx[tid] = 0;                 // zero padding => gather point 0
  {
    unsigned* xz = (unsigned*)&Xt[0][0];
    for (int i = tid; i < NSAMPLE * 288 / 2; i += 256) xz[i] = 0u;
  }
  __syncthreads();

  // ---- cylinder query: one wave, ordered first-64 compaction ----
  if (tid < 32) {
    const int lane = tid;
    const float r0 = sR[0], r1 = sR[1], r2 = sR[2];
    const float r3 = sR[3], r4 = sR[4], r5 = sR[5];
    const float r6 = sR[6], r7 = sR[7], r8 = sR[8];
    const float cx = sC[0], cy = sC[1], cz = sC[2];
    const float thresh = r2;                         // faithful to rot[...,2] shadowing bug
    const float* xb = xyz + (size_t)b * NPTS * 3;
    int cnt = 0;
    for (int n0 = 0; n0 < NPTS; n0 += 32) {
      if (cnt >= NSAMPLE) break;
      const int n = n0 + lane;
      const float dx = xb[3 * n + 0] - cx;
      const float dy = xb[3 * n + 1] - cy;
      const float dz = xb[3 * n + 2] - cz;
      const float t0 = dx * r0 + dy * r3 + dz * r6;
      const float t1 = dx * r1 + dy * r4 + dz * r7;
      const float t2 = dx * r2 + dy * r5 + dz * r8;
      const bool  m  = (t1 * t1 + t2 * t2 < thresh) && (t0 > HMIN_F) && (t0 < HMAX_F);
      const unsigned bal = (unsigned)__ballot(m);
      const int pos = cnt + __popc(bal & ((1u << lane) - 1u));
      if (m && pos < NSAMPLE) s_idx[pos] = n;
      cnt += __popc(bal);
    }
  }
  __syncthreads();

  // ---- build Xt: rotated recentred xyz (ch 0-2) + gathered features (ch 3-258) ----
  if (tid < NSAMPLE) {
    const int n = s_idx[tid];
    const float* xb = xyz + (size_t)b * NPTS * 3;
    const float dx = xb[3 * n + 0] - sC[0];
    const float dy = xb[3 * n + 1] - sC[1];
    const float dz = xb[3 * n + 2] - sC[2];
    Xt[tid][0] = f2bf(dx * sR[0] + dy * sR[3] + dz * sR[6]);
    Xt[tid][1] = f2bf(dx * sR[1] + dy * sR[4] + dz * sR[7]);
    Xt[tid][2] = f2bf(dx * sR[2] + dy * sR[5] + dz * sR[8]);
  }
  const float* fb = features + (size_t)b * CFEAT * NPTS;
  for (int i = tid; i < NSAMPLE * CFEAT; i += 256) {
    const int s = i & (NSAMPLE - 1);
    const int c = i >> 6;
    Xt[s][3 + c] = f2bf(fb[(size_t)c * NPTS + s_idx[s]]);
  }
  __syncthreads();

  const int wave = tid >> 5;
  const int lane = tid & 31;
  const int hi   = lane >> 4;
  const int nloc = lane & 15;

  // ---- layer 1: H1 = relu(W1 @ X + b1), 256x64, K=288 (padded) ----
  for (int tmi = 0; tmi < 2; ++tmi) {
    const int tm = wave * 2 + tmi;
    v8f acc[4];
#pragma unroll
    for (int tn = 0; tn < 4; ++tn) acc[tn] = (v8f){0, 0, 0, 0, 0, 0, 0, 0};
    for (int kc = 0; kc < 9; ++kc) {
      const v16bf a = load_a_frag(W1b, 288, tm * 16, kc * 32, lane);
#pragma unroll
      for (int tn = 0; tn < 4; ++tn) {
        const v16bf bfrag = load_b_frag(&Xt[0][0], 288, tn * 16, kc * 32, lane);
        acc[tn] = __builtin_amdgcn_wmma_f32_16x16x32_bf16(
            false, a, false, bfrag, (short)0, acc[tn], false, false);
      }
    }
    const int chB = tm * 16 + hi * 8;     // channels chB..chB+7 in this lane's VGPRs
    float bias[8];
#pragma unroll
    for (int r = 0; r < 8; ++r) bias[r] = b1[chB + r];
#pragma unroll
    for (int tn = 0; tn < 4; ++tn) {
      const int s = tn * 16 + nloc;
      float h[8];
#pragma unroll
      for (int r = 0; r < 8; ++r) h[r] = fmaxf(acc[tn][r] + bias[r], 0.0f);
      u32x4 st;
      st.x = pack2(h[0], h[1]); st.y = pack2(h[2], h[3]);
      st.z = pack2(h[4], h[5]); st.w = pack2(h[6], h[7]);
      *(u32x4*)&H1t[s][chB] = st;
    }
  }
  __syncthreads();

  // ---- layer 2: H2 = relu(W2 @ H1 + b2), then max over 64 samples ----
  for (int tmi = 0; tmi < 2; ++tmi) {
    const int tm = wave * 2 + tmi;
    v8f acc[4];
#pragma unroll
    for (int tn = 0; tn < 4; ++tn) acc[tn] = (v8f){0, 0, 0, 0, 0, 0, 0, 0};
    for (int kc = 0; kc < 8; ++kc) {
      const v16bf a = load_a_frag(W2b, 256, tm * 16, kc * 32, lane);
#pragma unroll
      for (int tn = 0; tn < 4; ++tn) {
        const v16bf bfrag = load_b_frag(&H1t[0][0], 256, tn * 16, kc * 32, lane);
        acc[tn] = __builtin_amdgcn_wmma_f32_16x16x32_bf16(
            false, a, false, bfrag, (short)0, acc[tn], false, false);
      }
    }
    const int chB = tm * 16 + hi * 8;
    float mx[8];
#pragma unroll
    for (int r = 0; r < 8; ++r) mx[r] = 0.0f;   // relu outputs are >= 0
#pragma unroll
    for (int tn = 0; tn < 4; ++tn) {
#pragma unroll
      for (int r = 0; r < 8; ++r) {
        float h = fmaxf(acc[tn][r] + b2[chB + r], 0.0f);
        // max over the 16 samples of this tile: lanes 0-15 and 16-31 reduce
        // independently (they hold different channel rows M=r vs M=r+8)
        h = fmaxf(h, __shfl_xor(h, 1));
        h = fmaxf(h, __shfl_xor(h, 2));
        h = fmaxf(h, __shfl_xor(h, 4));
        h = fmaxf(h, __shfl_xor(h, 8));
        mx[r] = fmaxf(mx[r], h);
      }
    }
    if (nloc == 0) {
#pragma unroll
      for (int r = 0; r < 8; ++r)
        out[((size_t)b * 256 + chB + r) * NPOINT + p] = mx[r];
    }
  }
}

// ---------------------------------------------------------------------------
extern "C" void kernel_launch(void* const* d_in, const int* in_sizes, int n_in,
                              void* d_out, int out_size, void* d_ws, size_t ws_size,
                              hipStream_t stream) {
  const float* xyz      = (const float*)d_in[0];
  const float* features = (const float*)d_in[1];
  const float* rot      = (const float*)d_in[2];
  const float* W1       = (const float*)d_in[3];
  const float* b1       = (const float*)d_in[4];
  const float* W2       = (const float*)d_in[5];
  const float* b2       = (const float*)d_in[6];
  float* out = (float*)d_out;

  char* ws = (char*)d_ws;
  int*            fps_idx = (int*)(ws);                              // 16384 B
  float*          new_xyz = (float*)(ws + 16384);                    // 49152 B
  unsigned short* W1b     = (unsigned short*)(ws + 65536);           // 147456 B
  unsigned short* W2b     = (unsigned short*)(ws + 65536 + 147456);  // 131072 B

  convert_weights_kernel<<<(256 * 288 + 255) / 256, 256, 0, stream>>>(W1, W2, W1b, W2b);
  fps_kernel<<<NBATCH, 256, 0, stream>>>(xyz, fps_idx, new_xyz);
  grasp_mlp_kernel<<<NBATCH * NPOINT, 256, 0, stream>>>(
      xyz, features, rot, b1, b2, new_xyz, W1b, W2b, out);
}
